// NystromAttention_42434276885035
// MI455X (gfx1250) — compile-verified, matrix operand order checked by reading, not run on hardware
//
#include <hip/hip_runtime.h>

// ---------------------------------------------------------------------------
// CDNA5 (gfx1250) Nystrom attention.  wave32, WMMA bf16 16x16x32, TDM staging.
// ---------------------------------------------------------------------------

typedef __attribute__((ext_vector_type(16))) __bf16 v16bf;
typedef __attribute__((ext_vector_type(8)))  float  v8f;
typedef __attribute__((ext_vector_type(4)))  unsigned v4u;
typedef __attribute__((ext_vector_type(8)))  int      v8i;
typedef __attribute__((ext_vector_type(4)))  int      v4i;

#define NB 8
#define SEQ 4096
#define FDIM 512
#define NH 8
#define AD 512
#define DH 64
#define NL 64
#define DKW 33
#define NSPLIT 8                 // S-splits for the flash kernel
#define NTILE (SEQ / NSPLIT / 64)  // 8 x 64-row tiles per split

__device__ inline v8f zero8() {
  v8f z = {0.f,0.f,0.f,0.f,0.f,0.f,0.f,0.f};
  return z;
}

__device__ inline v8f wmma_bf16(v16bf a, v16bf b, v8f c) {
  // D = A(16x32 bf16) x B(32x16 bf16) + C(16x16 f32)
  return __builtin_amdgcn_wmma_f32_16x16x32_bf16(false, a, false, b,
                                                 (short)0, c, false, false);
}

// Load a 16x32 bf16 fragment (A-style) from a row-major bf16 buffer.
// lane r = row within the 16-tile, hh = lane>>4 selects K sub-pattern.
// For the B operand, pass an n-major (transposed) buffer: rows = N, cols = K.
__device__ inline v16bf load_frag(const __bf16* base, int row, int pitch,
                                  int koff, int hh) {
  union { v16bf v; unsigned u[8]; } f;
  const __bf16* p = base + (size_t)row * pitch + koff;
#pragma unroll
  for (int j = 0; j < 8; ++j) {
    int k0 = ((j >> 2) << 4) + hh * 8 + ((j & 3) << 1);
    f.u[j] = *(const unsigned*)(p + k0);
  }
  return f.v;
}

__device__ inline float rmax16(float v) {
#pragma unroll
  for (int off = 1; off < 16; off <<= 1) v = fmaxf(v, __shfl_xor(v, off, 32));
  return v;
}
__device__ inline float rsum16(float v) {
#pragma unroll
  for (int off = 1; off < 16; off <<= 1) v += __shfl_xor(v, off, 32);
  return v;
}

// ---------------------------------------------------------------------------
// TDM: async DMA of a rows x cols bf16 2D tile (row pitch in elements) from
// global memory into LDS at byte offset lds_off.  D# layout per CDNA5 ISA ch8.
// Issue from ONE wave only (EXEC-ignored, per-wave TENSORcnt).
// ---------------------------------------------------------------------------
__device__ inline void tdm_load_tile_bf16(unsigned lds_off,
                                          const __bf16* gsrc, int rows,
                                          int cols, int pitch) {
  unsigned long long ga = (unsigned long long)(size_t)gsrc;
  v4u g0;
  g0[0] = 1u;                                        // count=1 user D#
  g0[1] = lds_off;                                   // lds_addr (bytes)
  g0[2] = (unsigned)(ga & 0xffffffffull);            // global_addr[31:0]
  g0[3] = (unsigned)((ga >> 32) & 0x01ffffffull)     // global_addr[56:32]
          | (2u << 30);                              // type=2 (image)
  v8i g1;
  g1[0] = (int)(1u << 16);                           // data_size=1 (2B/elem)
  g1[1] = (int)((unsigned)cols << 16);               // tensor_dim0[15:0]
  g1[2] = (int)(((unsigned)cols >> 16) |
                ((unsigned)SEQ << 16));              // td0 hi | tensor_dim1 lo
  g1[3] = (int)(((unsigned)SEQ >> 16) |
                ((unsigned)cols << 16));             // td1 hi | tile_dim0
  g1[4] = (int)((unsigned)rows & 0xffffu);           // tile_dim1 (tile_dim2=0)
  g1[5] = (int)(unsigned)pitch;                      // tensor_dim0_stride lo
  g1[6] = 0;                                         // stride0 hi|stride1 lo
  g1[7] = 0;                                         // stride1 hi
  v4i gz = {0, 0, 0, 0};
#if __has_include(<hip/amd_detail/amd_gfx1250_TDM.h>)
  v8i gz8 = {0, 0, 0, 0, 0, 0, 0, 0};
  __builtin_amdgcn_tensor_load_to_lds(g0, g1, gz, gz, gz8, 0);
#else
  __builtin_amdgcn_tensor_load_to_lds(g0, g1, gz, gz, 0);
#endif
}

// ---------------------------------------------------------------------------
// Kernel 1: fused QKV projection GEMM.  [32768 x 512] x [512 x 512] (+bias)
// Output bf16 in head-major layout [B,H,S,D], q/k scaled by D^-0.25.
// ---------------------------------------------------------------------------
__global__ __launch_bounds__(256) void qkv_gemm_kernel(
    const float* __restrict__ Xq, const float* __restrict__ Xk,
    const float* __restrict__ Xv, const float* __restrict__ Wq,
    const float* __restrict__ Wk, const float* __restrict__ Wv,
    const float* __restrict__ bq, const float* __restrict__ bk,
    const float* __restrict__ bv, __bf16* __restrict__ Oq,
    __bf16* __restrict__ Ok, __bf16* __restrict__ Ov) {
  const int z = blockIdx.z;
  const float* X = (z == 0) ? Xq : (z == 1) ? Xk : Xv;
  const float* W = (z == 0) ? Wq : (z == 1) ? Wk : Wv;
  const float* bias = (z == 0) ? bq : (z == 1) ? bk : bv;
  __bf16* O = (z == 0) ? Oq : (z == 1) ? Ok : Ov;
  const float scale = (z == 2) ? 1.0f : 0.35355339059327373f;  // 64^-0.25

  __shared__ __bf16 Alds[64 * 32];
  __shared__ __bf16 Btlds[64 * 32];  // n-major: Bt[n][k] = W[k][n]

  const int t = threadIdx.x;
  const int wave = t >> 5, lane = t & 31, hh = lane >> 4, r = lane & 15;
  const int mb = (wave >> 1) * 16, nbo = (wave & 1) * 32;
  const int gm0 = blockIdx.x * 64;
  const int gn0 = blockIdx.y * 64;

  v8f c0 = zero8(), c1 = zero8();
  for (int kk = 0; kk < FDIM; kk += 32) {
    __syncthreads();
#pragma unroll
    for (int i = 0; i < 8; ++i) {
      int li = t + i * 256;
      int row = li >> 5, col = li & 31;
      Alds[row * 32 + col] = (__bf16)X[(size_t)(gm0 + row) * FDIM + kk + col];
    }
#pragma unroll
    for (int i = 0; i < 8; ++i) {
      int li = t + i * 256;
      int k = li >> 6, n = li & 63;
      Btlds[n * 32 + k] = (__bf16)W[(size_t)(kk + k) * AD + gn0 + n];
    }
    __syncthreads();
    v16bf a  = load_frag(Alds, mb + r, 32, 0, hh);
    v16bf b0 = load_frag(Btlds, nbo + r, 32, 0, hh);
    v16bf b1 = load_frag(Btlds, nbo + 16 + r, 32, 0, hh);
    c0 = wmma_bf16(a, b0, c0);
    c1 = wmma_bf16(a, b1, c1);
  }
#pragma unroll
  for (int g = 0; g < 8; ++g) {
    int gm = gm0 + mb + hh * 8 + g;
    int bidx = gm >> 12, s = gm & (SEQ - 1);
#pragma unroll
    for (int tn = 0; tn < 2; ++tn) {
      int gn = gn0 + nbo + tn * 16 + r;
      float val = (((tn == 0) ? c0[g] : c1[g]) + bias[gn]) * scale;
      int h = gn >> 6, d = gn & 63;
      O[(((size_t)(bidx * NH + h)) * SEQ + s) * DH + d] = (__bf16)val;
    }
  }
}

// ---------------------------------------------------------------------------
// Kernel 2: landmark means.
// ---------------------------------------------------------------------------
__global__ void landmarks_kernel(const __bf16* __restrict__ q,
                                 const __bf16* __restrict__ k,
                                 __bf16* __restrict__ qland,
                                 __bf16* __restrict__ kland) {
  const __bf16* src = (blockIdx.y == 0) ? q : k;
  __bf16* dst = (blockIdx.y == 0) ? qland : kland;
  int bh = blockIdx.x >> 6, l = blockIdx.x & 63;
  int d = threadIdx.x;
  const __bf16* base = src + (size_t)bh * SEQ * DH;
  float acc = 0.f;
  for (int j = 0; j < 65; ++j) {
    int p = l * 65 + j;
    if (p >= SEQ) p -= SEQ;
    acc += (float)base[(size_t)p * DH + d];
  }
  dst[(size_t)blockIdx.x * DH + d] = (__bf16)(acc * (1.0f / 65.0f));
}

// ---------------------------------------------------------------------------
// Kernel 3: ker2 = softmax(q_land k_land^T) and Newton-Schulz pseudo-inverse.
// ---------------------------------------------------------------------------
__global__ __launch_bounds__(256) void ker2_inv_kernel(
    const __bf16* __restrict__ qland, const __bf16* __restrict__ kland,
    float* __restrict__ invout) {
  __shared__ float Km[64 * 64];
  __shared__ float Vm[64 * 64];
  __shared__ float Zm[64 * 64];
  __shared__ float Xm[64 * 64];
  const int t = threadIdx.x;
  const int bh = blockIdx.x;
  const __bf16* qL = qland + (size_t)bh * NL * DH;
  const __bf16* kL = kland + (size_t)bh * NL * DH;

  for (int p = 0; p < 16; ++p) {
    int idx = t + p * 256;
    int i = idx >> 6, j = idx & 63;
    float acc = 0.f;
    for (int dd = 0; dd < DH; ++dd)
      acc += (float)qL[i * DH + dd] * (float)kL[j * DH + dd];
    Km[idx] = acc;
  }
  __syncthreads();
  if (t < 64) {  // row softmax
    float m = -1e30f;
    for (int j = 0; j < 64; ++j) m = fmaxf(m, Km[t * 64 + j]);
    float ssum = 0.f;
    for (int j = 0; j < 64; ++j) {
      float e = __expf(Km[t * 64 + j] - m);
      Km[t * 64 + j] = e;
      ssum += e;
    }
    float is = 1.0f / ssum;
    for (int j = 0; j < 64; ++j) Km[t * 64 + j] *= is;
  }
  __syncthreads();
  if (t < 64) {  // column sums (scratch in Zm)
    float cs = 0.f;
    for (int i = 0; i < 64; ++i) cs += Km[i * 64 + t];
    Zm[t] = cs;
  }
  __syncthreads();
  if (t == 0) {
    float mx = 0.f;
    for (int j = 0; j < 64; ++j) mx = fmaxf(mx, Zm[j]);
    Zm[64] = 1.0f / mx;
  }
  __syncthreads();
  float dinv = Zm[64];
  for (int p = 0; p < 16; ++p) {
    int idx = t + p * 256;
    int i = idx >> 6, j = idx & 63;
    Vm[idx] = Km[j * 64 + i] * dinv;  // V = K^T / denom
  }
  __syncthreads();

  float reg[16];
  for (int iter = 0; iter < 6; ++iter) {
    for (int p = 0; p < 16; ++p) {  // Z = K@V
      int idx = t + p * 256;
      int i = idx >> 6, j = idx & 63;
      float a = 0.f;
      for (int kq = 0; kq < 64; ++kq) a += Km[i * 64 + kq] * Vm[kq * 64 + j];
      reg[p] = a;
    }
    __syncthreads();
    for (int p = 0; p < 16; ++p) Zm[t + p * 256] = reg[p];
    __syncthreads();
    for (int p = 0; p < 16; ++p) {  // X = 7Z - Z@Z
      int idx = t + p * 256;
      int i = idx >> 6, j = idx & 63;
      float a = 0.f;
      for (int kq = 0; kq < 64; ++kq) a += Zm[i * 64 + kq] * Zm[kq * 64 + j];
      reg[p] = 7.f * Zm[idx] - a;
    }
    __syncthreads();
    for (int p = 0; p < 16; ++p) Xm[t + p * 256] = reg[p];
    __syncthreads();
    for (int p = 0; p < 16; ++p) {  // X = 15Z - Z@X
      int idx = t + p * 256;
      int i = idx >> 6, j = idx & 63;
      float a = 0.f;
      for (int kq = 0; kq < 64; ++kq) a += Zm[i * 64 + kq] * Xm[kq * 64 + j];
      reg[p] = 15.f * Zm[idx] - a;
    }
    __syncthreads();
    for (int p = 0; p < 16; ++p) Xm[t + p * 256] = reg[p];
    __syncthreads();
    for (int p = 0; p < 16; ++p) {  // V = 0.25 (13V - V@X)
      int idx = t + p * 256;
      int i = idx >> 6, j = idx & 63;
      float a = 0.f;
      for (int kq = 0; kq < 64; ++kq) a += Vm[i * 64 + kq] * Xm[kq * 64 + j];
      reg[p] = 0.25f * (13.f * Vm[idx] - a);
    }
    __syncthreads();
    for (int p = 0; p < 16; ++p) Vm[t + p * 256] = reg[p];
    __syncthreads();
  }
  for (int p = 0; p < 16; ++p)
    invout[(size_t)bh * 4096 + t + p * 256] = Vm[t + p * 256];
}

// ---------------------------------------------------------------------------
// Kernel 4: partial flash:  softmax(q_land @ k^T) @ v over one S-chunk.
// grid (NSPLIT, B*H).  TDM double-buffers the k tile; v transposed manually.
// Emits unnormalized O plus per-row (m, l) for the split-combine.
// ---------------------------------------------------------------------------
__global__ __launch_bounds__(256) void ker3v_partial_kernel(
    const __bf16* __restrict__ qland, const __bf16* __restrict__ kbf,
    const __bf16* __restrict__ vbf, float* __restrict__ pOut,
    float* __restrict__ pM, float* __restrict__ pL) {
  __shared__ __bf16 kbuf[2][64 * 64];  // double-buffered k tile (TDM dest)
  __shared__ __bf16 vT[64 * 66];       // v tile transposed: vT[d][c]
  __shared__ __bf16 pbuf[64 * 64];     // probabilities for P@V
  const int t = threadIdx.x, wave = t >> 5, lane = t & 31;
  const int hh = lane >> 4, r = lane & 15;
  const int rb = wave >> 1, dhalf = wave & 1;
  const int bh = blockIdx.y, split = blockIdx.x;
  const int sbase = split * (SEQ / NSPLIT);
  const __bf16* qL = qland + (size_t)bh * NL * DH;
  const __bf16* kp = kbf + (size_t)bh * SEQ * DH;
  const __bf16* vp = vbf + (size_t)bh * SEQ * DH;

  v16bf aq0 = load_frag(qL, rb * 16 + r, DH, 0, hh);
  v16bf aq1 = load_frag(qL, rb * 16 + r, DH, 32, hh);

  v8f o0 = zero8(), o1 = zero8();
  float mrow[8], lrow[8];
#pragma unroll
  for (int g = 0; g < 8; ++g) { mrow[g] = -1e30f; lrow[g] = 0.f; }

  if (wave == 0)  // prime the pipeline: tile 0 -> kbuf[0]
    tdm_load_tile_bf16((unsigned)(size_t)&kbuf[0][0],
                       kp + (size_t)sbase * DH, 64, 64, DH);

  for (int it = 0; it < NTILE; ++it) {
    const int s0 = sbase + it * 64;
    __syncthreads();  // prior readers of vT / kbuf[(it+1)&1] are done
    if (wave == 0) {
      if (it + 1 < NTILE) {
        tdm_load_tile_bf16((unsigned)(size_t)&kbuf[(it + 1) & 1][0],
                           kp + (size_t)(s0 + 64) * DH, 64, 64, DH);
        __builtin_amdgcn_s_wait_tensorcnt(1);  // tile `it` landed (in-order)
      } else {
        __builtin_amdgcn_s_wait_tensorcnt(0);
      }
    }
#pragma unroll
    for (int i = 0; i < 16; ++i) {
      int li = t + i * 256;
      int c = li >> 6, d = li & 63;
      vT[d * 66 + c] = vp[(size_t)(s0 + c) * DH + d];
    }
    __syncthreads();

    const __bf16* kt = &kbuf[it & 1][0];
    v8f sc[4];
#pragma unroll
    for (int nt = 0; nt < 4; ++nt) {
      sc[nt] = zero8();
      sc[nt] = wmma_bf16(aq0, load_frag(kt, nt * 16 + r, 64, 0, hh), sc[nt]);
      sc[nt] = wmma_bf16(aq1, load_frag(kt, nt * 16 + r, 64, 32, hh), sc[nt]);
    }
#pragma unroll
    for (int g = 0; g < 8; ++g) {
      float tm = fmaxf(fmaxf(sc[0][g], sc[1][g]), fmaxf(sc[2][g], sc[3][g]));
      tm = rmax16(tm);
      float mnew = fmaxf(mrow[g], tm);
      float alpha = __expf(mrow[g] - mnew);
      float ts = 0.f;
#pragma unroll
      for (int nt = 0; nt < 4; ++nt) {
        float p = __expf(sc[nt][g] - mnew);
        sc[nt][g] = p;
        ts += p;
      }
      ts = rsum16(ts);
      lrow[g] = lrow[g] * alpha + ts;
      mrow[g] = mnew;
      o0[g] *= alpha;
      o1[g] *= alpha;
    }
    if (dhalf == 0) {
#pragma unroll
      for (int nt = 0; nt < 4; ++nt)
#pragma unroll
        for (int g = 0; g < 8; ++g)
          pbuf[(rb * 16 + hh * 8 + g) * 64 + nt * 16 + r] = (__bf16)sc[nt][g];
    }
    __syncthreads();
#pragma unroll
    for (int ks = 0; ks < 2; ++ks) {
      v16bf ap = load_frag(pbuf, rb * 16 + r, 64, ks * 32, hh);
      o0 = wmma_bf16(ap, load_frag(vT, dhalf * 32 + r, 66, ks * 32, hh), o0);
      o1 = wmma_bf16(ap, load_frag(vT, dhalf * 32 + 16 + r, 66, ks * 32, hh), o1);
    }
  }
  float* po = pOut + ((size_t)bh * NSPLIT + split) * (NL * DH);
#pragma unroll
  for (int g = 0; g < 8; ++g) {
    int row = rb * 16 + hh * 8 + g;
    po[row * DH + dhalf * 32 + r] = o0[g];
    po[row * DH + dhalf * 32 + 16 + r] = o1[g];
    if (dhalf == 0 && r == 0) {
      pM[((size_t)bh * NSPLIT + split) * NL + row] = mrow[g];
      pL[((size_t)bh * NSPLIT + split) * NL + row] = lrow[g];
    }
  }
}

// ---------------------------------------------------------------------------
// Kernel 4c: combine the NSPLIT partial flash results (log-sum-exp merge).
// ---------------------------------------------------------------------------
__global__ __launch_bounds__(256) void kv_combine_kernel(
    const float* __restrict__ pOut, const float* __restrict__ pM,
    const float* __restrict__ pL, float* __restrict__ kvout) {
  const int bh = blockIdx.x, t = threadIdx.x;
  for (int p = 0; p < 16; ++p) {
    int idx = t + p * 256;
    int row = idx >> 6;
    float mstar = -1e30f;
#pragma unroll
    for (int s = 0; s < NSPLIT; ++s)
      mstar = fmaxf(mstar, pM[((size_t)bh * NSPLIT + s) * NL + row]);
    float acc = 0.f, l = 0.f;
#pragma unroll
    for (int s = 0; s < NSPLIT; ++s) {
      float w = __expf(pM[((size_t)bh * NSPLIT + s) * NL + row] - mstar);
      acc += w * pOut[((size_t)bh * NSPLIT + s) * (NL * DH) + idx];
      l += w * pL[((size_t)bh * NSPLIT + s) * NL + row];
    }
    kvout[(size_t)bh * (NL * DH) + idx] = acc / l;
  }
}

// ---------------------------------------------------------------------------
// Kernel 4b: W2 = inv @ kv  (64x64 per (b,h)), output bf16 row-major [l][d].
// ---------------------------------------------------------------------------
__global__ __launch_bounds__(256) void w2_kernel(const float* __restrict__ invm,
                                                 const float* __restrict__ kv,
                                                 __bf16* __restrict__ W2) {
  const int bh = blockIdx.x, t = threadIdx.x;
  const float* iv = invm + (size_t)bh * 4096;
  const float* kp = kv + (size_t)bh * 4096;
  for (int p = 0; p < 16; ++p) {
    int idx = t + p * 256;
    int l = idx >> 6, d = idx & 63;
    float a = 0.f;
    for (int m = 0; m < 64; ++m) a += iv[l * 64 + m] * kp[m * 64 + d];
    W2[(size_t)bh * 4096 + idx] = (__bf16)a;
  }
}

// ---------------------------------------------------------------------------
// Kernel 5: x = softmax(q @ k_land^T) @ W2 + depthwise_conv(v)  -> bf16 [B,S,A]
// ---------------------------------------------------------------------------
__global__ __launch_bounds__(256) void ker1_conv_kernel(
    const __bf16* __restrict__ qbf, const __bf16* __restrict__ kland,
    const __bf16* __restrict__ W2, const __bf16* __restrict__ vbf,
    const float* __restrict__ convw, __bf16* __restrict__ xout) {
  __shared__ __bf16 w2T[64 * 64];   // w2T[d][l] = W2[l][d]
  __shared__ __bf16 vhal[96 * 64];  // v rows s0-16 .. s0+79
  __shared__ __bf16 pbuf[64 * 64];
  __shared__ float wconv[DKW];
  const int t = threadIdx.x, wave = t >> 5, lane = t & 31;
  const int hh = lane >> 4, r = lane & 15;
  const int rb = wave >> 1, dhalf = wave & 1;
  const int bh = blockIdx.y;
  const int s0 = blockIdx.x * 64;
  const int b = bh >> 3, h = bh & 7;
  const __bf16* qp = qbf + (size_t)bh * SEQ * DH;
  const __bf16* kL = kland + (size_t)bh * NL * DH;
  const __bf16* w2 = W2 + (size_t)bh * 4096;
  const __bf16* vp = vbf + (size_t)bh * SEQ * DH;

#pragma unroll
  for (int i = 0; i < 16; ++i) {
    int li = t + i * 256;
    int l = li >> 6, d = li & 63;
    w2T[d * 64 + l] = w2[li];
  }
  if (t < DKW) wconv[t] = convw[h * DKW + t];
  unsigned* vh32 = (unsigned*)vhal;
#pragma unroll
  for (int i = 0; i < 12; ++i) {
    int li = t + i * 256;  // 3072 dwords = 96x64 bf16
    int row = li >> 5, cu = li & 31;
    int srow = s0 - 16 + row;
    unsigned val = 0u;
    if (srow >= 0 && srow < SEQ)
      val = *(const unsigned*)(vp + (size_t)srow * DH + cu * 2);
    vh32[row * 32 + cu] = val;
  }
  __syncthreads();

  v16bf aq0 = load_frag(qp, s0 + rb * 16 + r, DH, 0, hh);
  v16bf aq1 = load_frag(qp, s0 + rb * 16 + r, DH, 32, hh);
  v8f sc[4];
#pragma unroll
  for (int nt = 0; nt < 4; ++nt) {
    sc[nt] = zero8();
    sc[nt] = wmma_bf16(aq0, load_frag(kL, nt * 16 + r, DH, 0, hh), sc[nt]);
    sc[nt] = wmma_bf16(aq1, load_frag(kL, nt * 16 + r, DH, 32, hh), sc[nt]);
  }
#pragma unroll
  for (int g = 0; g < 8; ++g) {  // full softmax over 64 landmark columns
    float m = fmaxf(fmaxf(sc[0][g], sc[1][g]), fmaxf(sc[2][g], sc[3][g]));
    m = rmax16(m);
    float ts = 0.f;
#pragma unroll
    for (int nt = 0; nt < 4; ++nt) {
      float p = __expf(sc[nt][g] - m);
      sc[nt][g] = p;
      ts += p;
    }
    ts = rsum16(ts);
    float is = 1.0f / ts;
#pragma unroll
    for (int nt = 0; nt < 4; ++nt) sc[nt][g] *= is;
  }
  if (dhalf == 0) {
#pragma unroll
    for (int nt = 0; nt < 4; ++nt)
#pragma unroll
      for (int g = 0; g < 8; ++g)
        pbuf[(rb * 16 + hh * 8 + g) * 64 + nt * 16 + r] = (__bf16)sc[nt][g];
  }
  __syncthreads();
  v8f o0 = zero8(), o1 = zero8();
#pragma unroll
  for (int ks = 0; ks < 2; ++ks) {
    v16bf ap = load_frag(pbuf, rb * 16 + r, 64, ks * 32, hh);
    o0 = wmma_bf16(ap, load_frag(w2T, dhalf * 32 + r, 64, ks * 32, hh), o0);
    o1 = wmma_bf16(ap, load_frag(w2T, dhalf * 32 + 16 + r, 64, ks * 32, hh), o1);
  }
#pragma unroll
  for (int g = 0; g < 8; ++g) {
    int rl = rb * 16 + hh * 8 + g;
    int s = s0 + rl;
#pragma unroll
    for (int tn = 0; tn < 2; ++tn) {
      int d = dhalf * 32 + tn * 16 + r;
      float cv = 0.f;
      for (int j = 0; j < DKW; ++j)
        cv += wconv[j] * (float)vhal[(rl + j) * 64 + d];
      float val = ((tn == 0) ? o0[g] : o1[g]) + cv;
      xout[((size_t)(b * SEQ + s)) * AD + h * DH + d] = (__bf16)val;
    }
  }
}

// ---------------------------------------------------------------------------
// Kernel 6: out = x @ Wo + bo.  [32768 x 512] x [512 x 512], f32 output.
// ---------------------------------------------------------------------------
__global__ __launch_bounds__(256) void outproj_kernel(
    const __bf16* __restrict__ X, const float* __restrict__ Wo,
    const float* __restrict__ bo, float* __restrict__ out) {
  __shared__ __bf16 Btlds[64 * 32];
  const int t = threadIdx.x, wave = t >> 5, lane = t & 31;
  const int hh = lane >> 4, r = lane & 15;
  const int mb = (wave >> 1) * 16, nbo = (wave & 1) * 32;
  const int gm0 = blockIdx.x * 64, gn0 = blockIdx.y * 64;
  v8f c0 = zero8(), c1 = zero8();
  for (int kk = 0; kk < AD; kk += 32) {
    __syncthreads();
#pragma unroll
    for (int i = 0; i < 8; ++i) {
      int li = t + i * 256;
      int k = li >> 6, n = li & 63;
      Btlds[n * 32 + k] = (__bf16)Wo[(size_t)(kk + k) * FDIM + gn0 + n];
    }
    __syncthreads();
    v16bf a = load_frag(X, gm0 + mb + r, AD, kk, hh);
    c0 = wmma_bf16(a, load_frag(Btlds, nbo + r, 32, 0, hh), c0);
    c1 = wmma_bf16(a, load_frag(Btlds, nbo + 16 + r, 32, 0, hh), c1);
  }
#pragma unroll
  for (int g = 0; g < 8; ++g) {
    int gm = gm0 + mb + hh * 8 + g;
#pragma unroll
    for (int tn = 0; tn < 2; ++tn) {
      int gn = gn0 + nbo + tn * 16 + r;
      out[(size_t)gm * FDIM + gn] = ((tn == 0) ? c0[g] : c1[g]) + bo[gn];
    }
  }
}

// ---------------------------------------------------------------------------
extern "C" void kernel_launch(void* const* d_in, const int* in_sizes, int n_in,
                              void* d_out, int out_size, void* d_ws,
                              size_t ws_size, hipStream_t stream) {
  (void)in_sizes; (void)n_in; (void)out_size; (void)ws_size;
  const float* query = (const float*)d_in[0];
  const float* key_  = (const float*)d_in[1];
  const float* value = (const float*)d_in[2];
  // d_in[3] = mask (all ones, unused by reference body)
  const float* Wq = (const float*)d_in[4];
  const float* bq = (const float*)d_in[5];
  const float* Wk = (const float*)d_in[6];
  const float* bk = (const float*)d_in[7];
  const float* Wv = (const float*)d_in[8];
  const float* bv = (const float*)d_in[9];
  const float* Wo = (const float*)d_in[10];
  const float* bo = (const float*)d_in[11];
  const float* convw = (const float*)d_in[12];
  float* out = (float*)d_out;

  char* ws = (char*)d_ws;
  size_t off = 0;
  auto alloc = [&](size_t bytes) -> void* {
    void* p = ws + off;
    off += (bytes + 255) & ~(size_t)255;
    return p;
  };
  const size_t NE = (size_t)NB * NH * SEQ * DH;  // 16,777,216
  __bf16* qbf   = (__bf16*)alloc(NE * 2);
  __bf16* kbf   = (__bf16*)alloc(NE * 2);
  __bf16* vbf   = (__bf16*)alloc(NE * 2);
  __bf16* qland = (__bf16*)alloc((size_t)NB * NH * NL * DH * 2);
  __bf16* kland = (__bf16*)alloc((size_t)NB * NH * NL * DH * 2);
  float*  invf  = (float*)alloc((size_t)NB * NH * NL * NL * 4);
  float*  kvf   = (float*)alloc((size_t)NB * NH * NL * DH * 4);
  __bf16* w2bf  = (__bf16*)alloc((size_t)NB * NH * NL * DH * 2);
  __bf16* xbf   = (__bf16*)alloc((size_t)NB * SEQ * AD * 2);
  float*  pO    = (float*)alloc((size_t)NB * NH * NSPLIT * NL * DH * 4);
  float*  pM    = (float*)alloc((size_t)NB * NH * NSPLIT * NL * 4);
  float*  pL    = (float*)alloc((size_t)NB * NH * NSPLIT * NL * 4);

  dim3 blk(256);
  qkv_gemm_kernel<<<dim3(512, 8, 3), blk, 0, stream>>>(
      query, key_, value, Wq, Wk, Wv, bq, bk, bv, qbf, kbf, vbf);
  landmarks_kernel<<<dim3(NB * NH * NL, 2), dim3(64), 0, stream>>>(
      qbf, kbf, qland, kland);
  ker2_inv_kernel<<<dim3(NB * NH), blk, 0, stream>>>(qland, kland, invf);
  ker3v_partial_kernel<<<dim3(NSPLIT, NB * NH), blk, 0, stream>>>(
      qland, kbf, vbf, pO, pM, pL);
  kv_combine_kernel<<<dim3(NB * NH), blk, 0, stream>>>(pO, pM, pL, kvf);
  w2_kernel<<<dim3(NB * NH), blk, 0, stream>>>(invf, kvf, w2bf);
  ker1_conv_kernel<<<dim3(SEQ / 64, NB * NH), blk, 0, stream>>>(
      qbf, kland, w2bf, vbf, convw, xbf);
  outproj_kernel<<<dim3(512, 8), blk, 0, stream>>>(xbf, Wo, bo, out);
}
